// MultiHeadAttention_5592047419547
// MI455X (gfx1250) — compile-verified, matrix operand order checked by reading, not run on hardware
//
#include <hip/hip_runtime.h>

// ---------------------------------------------------------------------------
// MI455X (gfx1250, wave32) fused causal MHA forward.
// All matmuls via v_wmma_f32_16x16x32_bf16 (fp32 in -> bf16 RNE, fp32 accum).
// bf16 tile staging uses GLOBAL_LOAD_ASYNC_TO_LDS_B128 (ASYNCcnt-tracked).
// GEMMs are double-buffered, single-barrier software pipelines.
// Workspace layout (d_ws), ~80 MB total:
//   [0)      qkv fp32 : 2048*6144*4  = 50331648 B
//   [qkv)    q  bf16  : 16*2048*128*2 = 8388608 B  (head-major)
//   [+8MB)   k  bf16  : 8388608 B                  (head-major)
//   [+16MB)  vT bf16  : 8388608 B  (vt[h][d][l])
//   [+24MB)  O  bf16  : 2048*2048*2 = 8388608 B    ([L][D])
// ---------------------------------------------------------------------------

typedef __bf16 bf16;
typedef __attribute__((ext_vector_type(8)))  bf16  v8bf;
typedef __attribute__((ext_vector_type(16))) bf16  v16bf;
typedef __attribute__((ext_vector_type(8)))  float v8f;
typedef __attribute__((ext_vector_type(4)))  int   v4i;

#if defined(__has_builtin)
#if __has_builtin(__builtin_amdgcn_global_load_async_to_lds_b128)
#define HAVE_ASYNC_LDS 1
#endif
#endif
#ifndef HAVE_ASYNC_LDS
#define HAVE_ASYNC_LDS 0
#endif

typedef __attribute__((address_space(1))) v4i* gas_v4i;
typedef __attribute__((address_space(3))) v4i* las_v4i;

// 16-byte global->LDS stage; async (ASYNCcnt) when available.
__device__ __forceinline__ void stage16(const void* g, void* l) {
#if HAVE_ASYNC_LDS
  __builtin_amdgcn_global_load_async_to_lds_b128(
      (gas_v4i)const_cast<void*>(g), (las_v4i)l, 0, 0);
#else
  *reinterpret_cast<uint4*>(l) = *reinterpret_cast<const uint4*>(g);
#endif
}

__device__ __forceinline__ void stage_wait() {
#if HAVE_ASYNC_LDS
#if __has_builtin(__builtin_amdgcn_s_wait_asynccnt)
  __builtin_amdgcn_s_wait_asynccnt(0);
#else
  asm volatile("s_wait_asynccnt 0" ::: "memory");
#endif
#endif
}

__device__ __forceinline__ unsigned short f2bf_raw(float f) {
  unsigned u = __builtin_bit_cast(unsigned, f);
  u += 0x7FFFu + ((u >> 16) & 1u);           // round-to-nearest-even
  return (unsigned short)(u >> 16);
}
__device__ __forceinline__ bf16 f2bf(float f) {
  return __builtin_bit_cast(bf16, f2bf_raw(f));
}
__device__ __forceinline__ unsigned long long pack4(float4 f) {
  return (unsigned long long)f2bf_raw(f.x) |
         ((unsigned long long)f2bf_raw(f.y) << 16) |
         ((unsigned long long)f2bf_raw(f.z) << 32) |
         ((unsigned long long)f2bf_raw(f.w) << 48);
}

// Load one WMMA 16x32 bf16 fragment (A-layout; B uses the same pattern on the
// transposed operand). Per the CDNA5 ISA table, lane half h takes
// K = 8h+0..7 (VGPR0-3) and 16+8h+0..7 (VGPR4-7).
__device__ __forceinline__ v16bf load_frag(const bf16* p, int h) {
  v8bf lo = *reinterpret_cast<const v8bf*>(p + 8 * h);
  v8bf hi = *reinterpret_cast<const v8bf*>(p + 16 + 8 * h);
  return __builtin_shufflevector(lo, hi, 0,1,2,3,4,5,6,7,8,9,10,11,12,13,14,15);
}

__device__ __forceinline__ v8f wmma_bf16(v16bf a, v16bf b, v8f c) {
  return __builtin_amdgcn_wmma_f32_16x16x32_bf16(
      /*neg_a=*/false, a, /*neg_b=*/false, b,
      /*c_mod=*/(short)0, c, /*reuse_a=*/false, /*reuse_b=*/false);
}

// ---------------------------------------------------------------------------
// GEMM: C[M][N] = A[M][K] * B[N][K]^T   (B row-major weights)
// Workgroup: 256 thr = 8 waves; tile 128(M) x 128(N); KC=32, double-buffered
// LDS (4 x 10KB), one barrier per stage. Wave grid 2(m) x 4(n); each wave
// computes 64x32 via 4x2 accumulators: 6 fragment loads feed 8 WMMAs/stage.
// Pipeline: commit tile n to LDS[buf] -> barrier -> issue tile n+1 loads
// (fp32->regs, or async-LDS into buf^1 for bf16 A) -> WMMAs on tile n.
// ---------------------------------------------------------------------------
template <bool A_BF16>
__global__ __launch_bounds__(256) void gemm_bf16_wmma(
    const void* __restrict__ Ap, const float* __restrict__ Bp,
    float* __restrict__ Cp, int M, int N, int K) {
  constexpr int TM = 128, TN = 128, KC = 32, LDA = 40, LDB = 40;
  __shared__ bf16 sA[2][TM * LDA];  // 2 x 10240 B
  __shared__ bf16 sB[2][TN * LDB];  // 2 x 10240 B

  const int tid = threadIdx.x;
  const int bm = blockIdx.y, bn = blockIdx.x;
  const int wave = tid >> 5, lane = tid & 31;
  const int wm = wave >> 2, wn = wave & 3;
  const int h = lane >> 4, ln = lane & 15;

  const float* Af = (const float*)Ap;
  const unsigned short* Ab = (const unsigned short*)Ap;

  float4 ra[4], rb[4];  // register staging for fp32 operands

  auto loadB = [&](int k0) {
#pragma unroll
    for (int p = 0; p < 4; ++p) {
      int id = p * 256 + tid;
      int r = id >> 3, c = (id & 7) * 4;
      const float* g = Bp + (size_t)(bn * TN + r) * K + k0 + c;
      if (k0 + KC < K) __builtin_prefetch(g + KC, 0, 0);
      rb[p] = *reinterpret_cast<const float4*>(g);
    }
  };
  auto loadA_f32 = [&](int k0) {
#pragma unroll
    for (int p = 0; p < 4; ++p) {
      int id = p * 256 + tid;
      int r = id >> 3, c = (id & 7) * 4;
      ra[p] = *reinterpret_cast<const float4*>(
          Af + (size_t)(bm * TM + r) * K + k0 + c);
    }
  };
  auto loadA_bf16_async = [&](int k0, int buf) {
#pragma unroll
    for (int p = 0; p < 2; ++p) {
      int id = p * 256 + tid;
      int r = id >> 2, c = (id & 3) * 8;
      stage16(Ab + (size_t)(bm * TM + r) * K + k0 + c, &sA[buf][r * LDA + c]);
    }
  };
  auto storeA_f32 = [&](int buf) {
#pragma unroll
    for (int p = 0; p < 4; ++p) {
      int id = p * 256 + tid;
      int r = id >> 3, c = (id & 7) * 4;
      *reinterpret_cast<unsigned long long*>(&sA[buf][r * LDA + c]) = pack4(ra[p]);
    }
  };
  auto storeB = [&](int buf) {
#pragma unroll
    for (int p = 0; p < 4; ++p) {
      int id = p * 256 + tid;
      int r = id >> 3, c = (id & 7) * 4;
      *reinterpret_cast<unsigned long long*>(&sB[buf][r * LDB + c]) = pack4(rb[p]);
    }
  };

  const v8f vzero = {0.f, 0.f, 0.f, 0.f, 0.f, 0.f, 0.f, 0.f};
  v8f acc[4][2];
#pragma unroll
  for (int ms = 0; ms < 4; ++ms)
#pragma unroll
    for (int ns = 0; ns < 2; ++ns) acc[ms][ns] = vzero;

  // prologue: tile 0 in flight
  if constexpr (A_BF16) loadA_bf16_async(0, 0);
  else loadA_f32(0);
  loadB(0);

  const int nIter = K / KC;
  for (int it = 0; it < nIter; ++it) {
    const int buf = it & 1;
    // commit tile `it` into LDS[buf]
    if constexpr (!A_BF16) storeA_f32(buf);
    storeB(buf);
    if constexpr (A_BF16) stage_wait();   // tile `it` async A landed
    __syncthreads();

    // launch tile `it+1` while tile `it` computes
    if (it + 1 < nIter) {
      const int k0n = (it + 1) * KC;
      if constexpr (A_BF16) loadA_bf16_async(k0n, buf ^ 1);
      else loadA_f32(k0n);
      loadB(k0n);
    }

    // 8 WMMAs on tile `it` (single kk step, KC==32)
    v16bf bfr[2];
#pragma unroll
    for (int ns = 0; ns < 2; ++ns)
      bfr[ns] = load_frag(&sB[buf][(wn * 32 + ns * 16 + ln) * LDB], h);
#pragma unroll
    for (int ms = 0; ms < 4; ++ms) {
      v16bf af = load_frag(&sA[buf][(wm * 64 + ms * 16 + ln) * LDA], h);
#pragma unroll
      for (int ns = 0; ns < 2; ++ns)
        acc[ms][ns] = wmma_bf16(af, bfr[ns], acc[ms][ns]);
    }
  }

  // ---- store C (C-layout: lane ln -> N, vgpr r + 8h -> M) ----
#pragma unroll
  for (int ms = 0; ms < 4; ++ms) {
    const int rbase = bm * TM + wm * 64 + ms * 16 + 8 * h;
#pragma unroll
    for (int ns = 0; ns < 2; ++ns) {
      const int coln = bn * TN + wn * 32 + ns * 16 + ln;
#pragma unroll
      for (int r = 0; r < 8; ++r)
        Cp[(size_t)(rbase + r) * N + coln] = acc[ms][ns][r];
    }
  }
}

// ---------------------------------------------------------------------------
// Per (l, head) row: RMSNorm(q,k) + RoPE(q,k), pack q/k head-major bf16,
// pack v transposed as vt[h][d][l]. One wave (32 lanes) per row, 4 elems/lane.
// ---------------------------------------------------------------------------
__global__ __launch_bounds__(32) void norm_rope_pack(
    const float* __restrict__ qkv,
    unsigned short* __restrict__ qb, unsigned short* __restrict__ kb,
    unsigned short* __restrict__ vtb) {
  const int l = blockIdx.x;        // 0..2047
  const int hh = blockIdx.y;       // 0..15
  const int lane = threadIdx.x;    // 0..31
  const int d0 = lane * 4;
  const size_t rowoff = (size_t)l * 6144 + (size_t)hh * 128;

  float4 q = *reinterpret_cast<const float4*>(qkv + rowoff + d0);
  float4 k = *reinterpret_cast<const float4*>(qkv + rowoff + 2048 + d0);
  float4 v = *reinterpret_cast<const float4*>(qkv + rowoff + 4096 + d0);

  float sq = q.x * q.x + q.y * q.y + q.z * q.z + q.w * q.w;
  float sk = k.x * k.x + k.y * k.y + k.z * k.z + k.w * k.w;
#pragma unroll
  for (int m = 1; m < 32; m <<= 1) {
    sq += __shfl_xor(sq, m, 32);
    sk += __shfl_xor(sk, m, 32);
  }
  const float rq = rsqrtf(sq * (1.0f / 128.0f) + 1e-5f);
  const float rk = rsqrtf(sk * (1.0f / 128.0f) + 1e-5f);

  // RoPE: pairs p0 = 2*lane, p1 = 2*lane+1; inv_freq = base^(-2p/128)
  const float LOGB = 11.512925464970229f;  // ln(100000)
  const float lf = (float)l;
  const int p0 = lane * 2;
  float a0 = lf * __expf(-(2.0f * (float)p0 / 128.0f) * LOGB);
  float a1 = lf * __expf(-(2.0f * (float)(p0 + 1) / 128.0f) * LOGB);
  float c0 = __cosf(a0), s0 = __sinf(a0);
  float c1 = __cosf(a1), s1 = __sinf(a1);

  // reference: out[2p] = odd*cos - even*sin ; out[2p+1] = odd*sin + even*cos
  float qe0 = q.x * rq, qo0 = q.y * rq, qe1 = q.z * rq, qo1 = q.w * rq;
  float ke0 = k.x * rk, ko0 = k.y * rk, ke1 = k.z * rk, ko1 = k.w * rk;
  float qy0 = qo0 * c0 - qe0 * s0, qy1 = qo0 * s0 + qe0 * c0;
  float qy2 = qo1 * c1 - qe1 * s1, qy3 = qo1 * s1 + qe1 * c1;
  float ky0 = ko0 * c0 - ke0 * s0, ky1 = ko0 * s0 + ke0 * c0;
  float ky2 = ko1 * c1 - ke1 * s1, ky3 = ko1 * s1 + ke1 * c1;

  const size_t qkoff = ((size_t)hh * 2048 + l) * 128 + d0;
  uint2 pq, pk2;
  pq.x = (unsigned)f2bf_raw(qy0) | ((unsigned)f2bf_raw(qy1) << 16);
  pq.y = (unsigned)f2bf_raw(qy2) | ((unsigned)f2bf_raw(qy3) << 16);
  pk2.x = (unsigned)f2bf_raw(ky0) | ((unsigned)f2bf_raw(ky1) << 16);
  pk2.y = (unsigned)f2bf_raw(ky2) | ((unsigned)f2bf_raw(ky3) << 16);
  *reinterpret_cast<uint2*>(qb + qkoff) = pq;
  *reinterpret_cast<uint2*>(kb + qkoff) = pk2;

  // V transposed: vt[h][d][l]
  const size_t vbase = ((size_t)hh * 128 + d0) * 2048 + l;
  vtb[vbase + 0 * 2048] = f2bf_raw(v.x);
  vtb[vbase + 1 * 2048] = f2bf_raw(v.y);
  vtb[vbase + 2 * 2048] = f2bf_raw(v.z);
  vtb[vbase + 3 * 2048] = f2bf_raw(v.w);
}

// ---------------------------------------------------------------------------
// Flash attention, causal. Block = 128 thr = 4 waves; 64 q-rows per block
// (16 rows/wave); j-tiles of 64 keys; hd = 128 (8 accumulators of 16x16).
// ---------------------------------------------------------------------------
__global__ __launch_bounds__(128) void flash_attn(
    const unsigned short* __restrict__ qb, const unsigned short* __restrict__ kb,
    const unsigned short* __restrict__ vtb, unsigned short* __restrict__ ob) {
  constexpr int L = 2048, HD = 128, LQK = 136, LVT = 72, LP = 72;
  __shared__ bf16 sQ[64 * LQK];       // 17408 B
  __shared__ bf16 sK[64 * LQK];       // 17408 B
  __shared__ bf16 sVT[128 * LVT];     // 18432 B
  __shared__ bf16 sP[4 * 16 * LP];    //  9216 B  (per-wave P stage)

  const int qt = blockIdx.x;          // q tile 0..31
  const int hh = blockIdx.y;          // head 0..15
  const int tid = threadIdx.x;
  const int wave = tid >> 5, lane = tid & 31;
  const int h = lane >> 4, ln = lane & 15;
  const size_t hoff = (size_t)hh * L * HD;

  // stage Q tile (64 x 128 bf16); completion folded into first loop wait
#pragma unroll
  for (int p = 0; p < 8; ++p) {
    int chunk = p * 128 + tid;
    int r = chunk >> 4;
    int c = (chunk & 15) * 8;
    stage16(qb + hoff + (size_t)(qt * 64 + r) * HD + c, &sQ[r * LQK + c]);
  }

  const v8f vzero = {0.f, 0.f, 0.f, 0.f, 0.f, 0.f, 0.f, 0.f};
  float mrow[8], lrow[8];
  v8f accO[8];
#pragma unroll
  for (int i = 0; i < 8; ++i) {
    mrow[i] = -__builtin_inff();
    lrow[i] = 0.f;
    accO[i] = vzero;
  }
  bf16* sPw = &sP[wave * 16 * LP];
  const float sc = 0.08838834764831845f;  // 1/sqrt(128)

  for (int jt = 0; jt <= qt; ++jt) {
    __syncthreads();
    // stage K tile (64 x 128) and V^T tile (128 x 64)
#pragma unroll
    for (int p = 0; p < 8; ++p) {
      int chunk = p * 128 + tid;
      int r = chunk >> 4;
      int c = (chunk & 15) * 8;
      stage16(kb + hoff + (size_t)(jt * 64 + r) * HD + c, &sK[r * LQK + c]);
    }
#pragma unroll
    for (int p = 0; p < 8; ++p) {
      int chunk = p * 128 + tid;
      int d = chunk >> 3;
      int c = (chunk & 7) * 8;
      stage16(vtb + (size_t)(hh * HD + d) * L + jt * 64 + c, &sVT[d * LVT + c]);
    }
    stage_wait();
    __syncthreads();

    // S = Qw (16x128) @ K^T -> 4 x (16x16)
    v8f s[4] = {vzero, vzero, vzero, vzero};
#pragma unroll
    for (int kk = 0; kk < 128; kk += 32) {
      v16bf aq = load_frag(&sQ[(wave * 16 + ln) * LQK + kk], h);
#pragma unroll
      for (int sb = 0; sb < 4; ++sb) {
        v16bf bk = load_frag(&sK[(sb * 16 + ln) * LQK + kk], h);
        s[sb] = wmma_bf16(aq, bk, s[sb]);
      }
    }

    // online softmax (rows r+8h live across the 16-lane half; butterfly <16)
    const bool diag = (jt == qt);
    float alpha[8];
#pragma unroll
    for (int r = 0; r < 8; ++r) {
      const int rowl = wave * 16 + r + 8 * h;  // local row in 64-row q tile
      float mx = -__builtin_inff();
#pragma unroll
      for (int sb = 0; sb < 4; ++sb) {
        float v = s[sb][r] * sc;
        if (diag && (sb * 16 + ln) > rowl) v = -__builtin_inff();
        s[sb][r] = v;
        mx = fmaxf(mx, v);
      }
#pragma unroll
      for (int m = 1; m < 16; m <<= 1) mx = fmaxf(mx, __shfl_xor(mx, m, 32));
      const float mnew = fmaxf(mrow[r], mx);
      alpha[r] = __expf(mrow[r] - mnew);
      float sum = 0.f;
#pragma unroll
      for (int sb = 0; sb < 4; ++sb) {
        float pe = __expf(s[sb][r] - mnew);
        sum += pe;
        sPw[(r + 8 * h) * LP + sb * 16 + ln] = f2bf(pe);
      }
#pragma unroll
      for (int m = 1; m < 16; m <<= 1) sum += __shfl_xor(sum, m, 32);
      lrow[r] = lrow[r] * alpha[r] + sum;
      mrow[r] = mnew;
    }
    // rescale O accumulators
#pragma unroll
    for (int db = 0; db < 8; ++db)
#pragma unroll
      for (int r = 0; r < 8; ++r) accO[db][r] = accO[db][r] * alpha[r];

    // O += P (16x64) @ V (64x128); V^T rows give contiguous b128 B-frags
#pragma unroll
    for (int kk = 0; kk < 64; kk += 32) {
      v16bf ap = load_frag(&sPw[ln * LP + kk], h);
#pragma unroll
      for (int db = 0; db < 8; ++db) {
        v16bf bv = load_frag(&sVT[(db * 16 + ln) * LVT + kk], h);
        accO[db] = wmma_bf16(ap, bv, accO[db]);
      }
    }
  }

  // O /= l ; write merged-head bf16 [L][D]
#pragma unroll
  for (int db = 0; db < 8; ++db) {
    const int d = db * 16 + ln;
#pragma unroll
    for (int r = 0; r < 8; ++r) {
      const int row = qt * 64 + wave * 16 + r + 8 * h;
      const float o = accO[db][r] / lrow[r];
      ob[(size_t)row * 2048 + (size_t)hh * HD + d] = f2bf_raw(o);
    }
  }
}

// ---------------------------------------------------------------------------
extern "C" void kernel_launch(void* const* d_in, const int* in_sizes, int n_in,
                              void* d_out, int out_size, void* d_ws, size_t ws_size,
                              hipStream_t stream) {
  (void)in_sizes; (void)n_in; (void)out_size; (void)ws_size;
  const float* x     = (const float*)d_in[0];
  const float* W_qkv = (const float*)d_in[1];
  const float* W_out = (const float*)d_in[2];
  // d_in[3] (block_mask) unused: causality is computed analytically.

  char* ws = (char*)d_ws;
  float* qkv_ws = (float*)ws;  // 2048*6144 fp32
  unsigned short* qb  = (unsigned short*)(ws + (size_t)2048 * 6144 * 4);
  unsigned short* kb  = qb  + (size_t)16 * 2048 * 128;
  unsigned short* vtb = kb  + (size_t)16 * 2048 * 128;
  unsigned short* ob  = vtb + (size_t)16 * 2048 * 128;
  float* out = (float*)d_out;

  dim3 g1(6144 / 128, 2048 / 128);
  gemm_bf16_wmma<false><<<g1, 256, 0, stream>>>(x, W_qkv, qkv_ws, 2048, 6144, 2048);

  norm_rope_pack<<<dim3(2048, 16), 32, 0, stream>>>(qkv_ws, qb, kb, vtb);

  flash_attn<<<dim3(32, 16), 128, 0, stream>>>(qb, kb, vtb, ob);

  dim3 g4(2048 / 128, 2048 / 128);
  gemm_bf16_wmma<true><<<g4, 256, 0, stream>>>(ob, W_out, out, 2048, 2048, 2048);
}
